// PlasticActorCritic_25872882992034
// MI455X (gfx1250) — matrix operand order repaired and checked by reading, not compiled, not run on hardware
//
#include <hip/hip_runtime.h>
#include <math.h>
#include <stdint.h>

// Problem sizes (fixed by the reference)
#define T_STEPS 4096
#define IN_DIM  512
#define H_DIM   2048
#define OUT_DIM 4

// Persistent scan kernel geometry: 128 WGs x 512 threads = 65536 threads,
// exactly one (gate,chunk,column) dot-product task per thread per phase.
#define NWG 128
#define BLK 512

typedef _Float16 v16h __attribute__((ext_vector_type(16)));
typedef _Float16 v8h  __attribute__((ext_vector_type(8)));
typedef float    v8f  __attribute__((ext_vector_type(8)));

__device__ __forceinline__ float sigm(float x) { return 1.f / (1.f + __expf(-x)); }

__device__ __forceinline__ unsigned long long sm64(unsigned long long z) {
  z += 0x9E3779B97F4A7C15ull;
  z = (z ^ (z >> 30)) * 0xBF58476D1CE4E5B9ull;
  z = (z ^ (z >> 27)) * 0x94D049BB133111EBull;
  return z ^ (z >> 31);
}

// Grid-wide barrier: monotonically increasing counter, uniform control flow,
// release fence before arrive, acquire fence after release is observed.
__device__ __forceinline__ void gbar(unsigned* cnt, unsigned expect) {
  __syncthreads();
  if (threadIdx.x == 0) {
    __threadfence();
    atomicAdd(cnt, 1u);
    volatile unsigned* vc = cnt;
    while (*vc < expect) __builtin_amdgcn_s_sleep(2);
  }
  __syncthreads();
  __threadfence();
}

// ---------------------------------------------------------------------------
// Convert inputs to f16 staging, copy hebb0, init h ping-pong + barrier cnt.
// ---------------------------------------------------------------------------
__global__ void k_convert(const float* __restrict__ x,
                          const float* __restrict__ Wz, const float* __restrict__ Wr,
                          const float* __restrict__ Wh,
                          const float* __restrict__ Uz, const float* __restrict__ Ur,
                          const float* __restrict__ Uh, const float* __restrict__ Al,
                          const float* __restrict__ hebb0, const float* __restrict__ h0,
                          _Float16* __restrict__ X16, _Float16* __restrict__ W16,
                          _Float16* __restrict__ U16,
                          float* __restrict__ hebb, float* __restrict__ hbuf,
                          unsigned* __restrict__ cnt) {
  const long long NX = (long long)T_STEPS * IN_DIM;
  const long long NW = 3LL * IN_DIM * H_DIM;
  const long long NH = (long long)H_DIM * H_DIM;
  const long long NU = 4LL * NH;
  const long long TOT = NX + NW + NU + NH + H_DIM + 1;
  const long long stride = (long long)gridDim.x * blockDim.x;
  for (long long i = (long long)blockIdx.x * blockDim.x + threadIdx.x; i < TOT; i += stride) {
    long long r = i;
    if (r < NX) { X16[r] = (_Float16)x[r]; continue; }
    r -= NX;
    if (r < NW) {
      long long m = r / ((long long)IN_DIM * H_DIM);
      long long o = r % ((long long)IN_DIM * H_DIM);
      const float* s = (m == 0) ? Wz : ((m == 1) ? Wr : Wh);
      W16[r] = (_Float16)s[o];
      continue;
    }
    r -= NW;
    if (r < NU) {
      long long m = r / NH, o = r % NH;
      const float* s = (m == 0) ? Uz : ((m == 1) ? Ur : ((m == 2) ? Uh : Al));
      U16[r] = (_Float16)s[o];
      continue;
    }
    r -= NU;
    if (r < NH) { hebb[r] = hebb0[r]; continue; }
    r -= NH;
    if (r < H_DIM) { hbuf[r] = h0[r]; hbuf[H_DIM + r] = 0.f; continue; }
    *cnt = 0u;
  }
}

// ---------------------------------------------------------------------------
// WMMA f16 GEMM: P[b] = f16( x @ W[b] + bias[b] ), b in {z,r,h}.
// Block = 8 waves; each wave owns a 16x16 C tile.
// B tile is staged into LDS in per-lane *fragment order* so each lane reads
// its 16 B-operand halfs as two contiguous, 16B-aligned ds_load_b128s
// (element (k,n) -> sB[((k>>4)*16 + n)*24 + (k&15)]; 48B row stride keeps
// b128 alignment and spreads banks, stride-12 -> 16 distinct banks/half-wave).
// ---------------------------------------------------------------------------
__global__ void __launch_bounds__(256) k_gemm(const _Float16* __restrict__ X16,
                                              const _Float16* __restrict__ W16,
                                              const float* __restrict__ bz,
                                              const float* __restrict__ br,
                                              const float* __restrict__ bh,
                                              _Float16* __restrict__ P16) {
  const int nt = blockIdx.x;        // N tile (16 cols)
  const int mb = blockIdx.y;        // 128-row M block
  const int b  = blockIdx.z;        // which weight matrix
  const float* bias = (b == 0) ? bz : ((b == 1) ? br : bh);
  const _Float16* W = W16 + (size_t)b * IN_DIM * H_DIM;
  _Float16* P = P16 + (size_t)b * T_STEPS * H_DIM;

  const int tid  = threadIdx.x;
  const int wave = tid >> 5;
  const int lane = tid & 31;
  const int hi   = lane >> 4;       // half-wave select
  const int lm   = lane & 15;
  const int arow = mb * 128 + wave * 16 + lm;

  __shared__ _Float16 sB[32 * 24];  // 32 lanes x (16 frag halfs + 8 pad)
  v8f acc = {};

  for (int k0 = 0; k0 < IN_DIM; k0 += 32) {
    __syncthreads();
    {  // cooperative B-tile stage: each thread one dword (2 halfs -> 2 lanes)
      const int kr = tid >> 3;           // k row 0..31
      const int cp = (tid & 7) * 2;      // even col
      const _Float16* src = W + (size_t)(k0 + kr) * H_DIM + nt * 16 + cp;
      const uint32_t d = *(const uint32_t*)src;
      const int lbase = (kr >> 4) * 16;  // lane group by k-half
      const int e = kr & 15;             // fragment element
      sB[(lbase + cp) * 24 + e]     = ((const _Float16*)&d)[0];
      sB[(lbase + cp + 1) * 24 + e] = ((const _Float16*)&d)[1];
      if (k0 + 32 < IN_DIM)
        __builtin_prefetch(src + 32 * H_DIM, 0, 0);  // global_prefetch_b8
    }
    __syncthreads();

    // A fragment (contiguous k order, matching B): elems 0..7 <-> K=hi*16+e,
    // elems 8..15 <-> K=hi*16+8+e, row = lane%16.  Two global b128 loads.
    v16h a, bf;
    const _Float16* ap = X16 + (size_t)arow * IN_DIM + k0 + hi * 16;
    v8h alo = *(const v8h*)ap;
    v8h ahi = *(const v8h*)(ap + 8);

    // B fragment: 32 contiguous bytes in per-lane order -> 2x ds_load_b128
    const v8h* bp = (const v8h*)&sB[lane * 24];
    v8h blo = bp[0];
    v8h bhi = bp[1];
#pragma unroll
    for (int e = 0; e < 8; ++e) {
      a[e] = alo[e]; a[8 + e] = ahi[e];
      bf[e] = blo[e]; bf[8 + e] = bhi[e];
    }

    acc = __builtin_amdgcn_wmma_f32_16x16x32_f16(false, a, false, bf,
                                                 (short)0, acc, false, false);
  }

  // C layout: VGPR v, half-wave hi -> row = v + hi*8, col = lane%16
  const int col = nt * 16 + lm;
  const float bv = bias[col];
#pragma unroll
  for (int v = 0; v < 8; ++v) {
    const int row = mb * 128 + wave * 16 + hi * 8 + v;
    P[(size_t)row * H_DIM + col] = (_Float16)(acc[v] + bv);
  }
}

// ---------------------------------------------------------------------------
// Persistent plastic-GRU scan. 3 grid barriers / step:
//   A: partial dots for z,r gates (h_t resident via LDS chunk)
//   B: per-WG rh chunk in LDS; h_tilde dots fused with lazy Hebbian update
//      (hebb read once, decayed + outer-product, written back, consumed)
//   F: finalize z / h_tilde / h_{t+1}; deterministic y@Wo partial reduction
// ---------------------------------------------------------------------------
__global__ void __launch_bounds__(BLK) k_scan(
    const _Float16* __restrict__ projZ, const _Float16* __restrict__ projR,
    const _Float16* __restrict__ projH,
    const _Float16* __restrict__ Uz16, const _Float16* __restrict__ Ur16,
    const _Float16* __restrict__ Uh16, const _Float16* __restrict__ Al16,
    float* __restrict__ hebb, float* __restrict__ hbuf,
    float* __restrict__ partZ, float* __restrict__ partR, float* __restrict__ partB,
    float* __restrict__ outPart, const float* __restrict__ Wo,
    const float* __restrict__ etap, unsigned* __restrict__ cnt) {
  const int w   = blockIdx.x;
  const int tid = threadIdx.x;
  const float eta   = etap[0];
  const float ometa = 1.f - eta;
  __shared__ float smem[272];
  unsigned expect = 0;

  // Phase A mapping (WG-uniform gate/chunk): tg = w*512+tid, q = tg>>11
  const int qa = w >> 2;
  const int ga = qa >> 4;                 // 0 = z gate, 1 = r gate
  const int ca = qa & 15;                 // 128-row i-chunk
  const int ja = (w & 3) * 512 + tid;     // output column
  const _Float16* Ua = (ga == 0) ? Uz16 : Ur16;
  float* partA = (ga == 0) ? partZ : partR;

  // Phase B mapping: WG owns i-chunk cb and 256 columns; 2 threads/column
  const int cb    = w & 15;
  const int jB    = (w >> 4) * 256 + (tid & 255);
  const int ihalf = tid >> 8;

  for (int t = 0; t < T_STEPS; ++t) {
    const float* hc  = hbuf + (t & 1) * H_DIM;        // h_t
    float*       hnb = hbuf + ((t & 1) ^ 1) * H_DIM;  // h_{t+1} slot
    const float* hp  = hnb;                           // currently holds h_{t-1}

    // ---------------- Phase A: z/r partial dots ----------------
    if (tid < 128) smem[tid] = hc[ca * 128 + tid];
    __syncthreads();
    {
      const _Float16* col = Ua + (size_t)(ca * 128) * H_DIM + ja;
      float acc = 0.f;
#pragma unroll 8
      for (int k = 0; k < 128; ++k)
        acc += smem[k] * (float)col[(size_t)k * H_DIM];
      partA[(size_t)ca * H_DIM + ja] = acc;
    }
    expect += NWG; gbar(cnt, expect);

    // ---------------- Phase B: h_tilde dots + lazy hebb update ----------------
    if (tid < 128) {
      const int i = cb * 128 + tid;
      float s = (float)projR[(size_t)t * H_DIM + i];
#pragma unroll
      for (int q = 0; q < 16; ++q) s += partR[(size_t)q * H_DIM + i];
      const float r = sigm(s);
      smem[tid]       = r * hc[i];   // rh chunk
      smem[128 + tid] = hp[i];       // h_{t-1} chunk (update row factor)
    }
    __syncthreads();
    {
      const size_t rowb = (size_t)(cb * 128 + ihalf * 64) * H_DIM + jB;
      const _Float16* uhp = Uh16 + rowb;
      const _Float16* alp = Al16 + rowb;
      float* hbp = hebb + rowb;
      const float hcj = hc[jB];
      const int sb = ihalf * 64;
      float acc = 0.f;
      if (t > 0) {
#pragma unroll 4
        for (int k = 0; k < 64; ++k) {
          const size_t off = (size_t)k * H_DIM;
          float hv = hbp[off];
          hv = ometa * hv + eta * smem[128 + sb + k] * hcj;  // hebb_t element
          hbp[off] = hv;
          acc += smem[sb + k] * ((float)uhp[off] + (float)alp[off] * hv);
        }
      } else {  // first step uses hebb0 as-is, no pending update
#pragma unroll 4
        for (int k = 0; k < 64; ++k) {
          const size_t off = (size_t)k * H_DIM;
          acc += smem[sb + k] * ((float)uhp[off] + (float)alp[off] * hbp[off]);
        }
      }
      partB[(size_t)(cb * 2 + ihalf) * H_DIM + jB] = acc;
    }
    expect += NWG; gbar(cnt, expect);

    // ---------------- Phase F: finalize h_{t+1}, y@Wo partials ----------------
    if (w < 4) {
      const int j = w * 512 + tid;
      float sz = (float)projZ[(size_t)t * H_DIM + j];
#pragma unroll
      for (int q = 0; q < 16; ++q) sz += partZ[(size_t)q * H_DIM + j];
      const float z = sigm(sz);
      float sh = (float)projH[(size_t)t * H_DIM + j];
#pragma unroll
      for (int q = 0; q < 32; ++q) sh += partB[(size_t)q * H_DIM + j];
      const float htl = tanhf(sh);
      const float hn = (1.f - z) * hc[j] + z * htl;
      hnb[j] = hn;
      const int lane = tid & 31, wid = tid >> 5;
#pragma unroll
      for (int o = 0; o < OUT_DIM; ++o) {
        float v = hn * Wo[(size_t)j * OUT_DIM + o];
        for (int off = 16; off > 0; off >>= 1) v += __shfl_down(v, off, 32);
        if (lane == 0) smem[256 + wid] = v;
        __syncthreads();
        if (tid == 0) {
          float s = 0.f;
          for (int kk = 0; kk < 16; ++kk) s += smem[256 + kk];
          outPart[((size_t)t * OUT_DIM + o) * 4 + w] = s;
        }
        __syncthreads();
      }
    }
    expect += NWG; gbar(cnt, expect);
  }
}

// ---------------------------------------------------------------------------
// Apply the final pending Hebbian update to produce hebbT; copy hT.
// hbuf[0] = h_T (T even), hbuf[1] = h_{T-1}.
// ---------------------------------------------------------------------------
__global__ void k_fhebb(const float* __restrict__ hebb, const float* __restrict__ hbuf,
                        const float* __restrict__ etap,
                        float* __restrict__ outHebb, float* __restrict__ outHT) {
  const long long i = (long long)blockIdx.x * blockDim.x + threadIdx.x;
  const long long NH = (long long)H_DIM * H_DIM;
  if (i < NH) {
    const float eta = etap[0];
    const long long r = i >> 11, c = i & (H_DIM - 1);
    outHebb[i] = (1.f - eta) * hebb[i] + eta * hbuf[H_DIM + r] * hbuf[c];
  }
  if (i < H_DIM) outHT[i] = hbuf[i];
}

// ---------------------------------------------------------------------------
// Combine y@Wo partials with bias; deterministic Gumbel-argmax action.
// ---------------------------------------------------------------------------
__global__ void k_post(const float* __restrict__ outPart, const float* __restrict__ bo,
                       const int* __restrict__ seedp,
                       float* __restrict__ out, int* __restrict__ act) {
  const int t = blockIdx.x * blockDim.x + threadIdx.x;
  if (t >= T_STEPS) return;
  float v[OUT_DIM];
#pragma unroll
  for (int o = 0; o < OUT_DIM; ++o) {
    const float* p = outPart + ((size_t)t * OUT_DIM + o) * 4;
    float s = bo[o] + p[0] + p[1] + p[2] + p[3];
    out[t * OUT_DIM + o] = s;
    v[o] = s;
  }
  const unsigned long long base =
      ((unsigned long long)(unsigned)seedp[0] << 32) ^ 0x9E3779B97F4A7C15ull;
  float best = -1e30f; int bi = 0;
#pragma unroll
  for (int o = 0; o < 2; ++o) {
    const unsigned long long h = sm64(base + (unsigned long long)t * 2ull + o);
    const float u = (float)((h >> 40) + 1ull) * (1.f / 16777218.f);
    const float g = -logf(-logf(u));
    const float s = v[o] + g;
    if (s > best) { best = s; bi = o; }
  }
  act[t] = bi;
}

// ---------------------------------------------------------------------------
extern "C" void kernel_launch(void* const* d_in, const int* in_sizes, int n_in,
                              void* d_out, int out_size, void* d_ws, size_t ws_size,
                              hipStream_t stream) {
  (void)in_sizes; (void)n_in; (void)out_size; (void)ws_size;
  const float* x     = (const float*)d_in[0];
  const float* h0    = (const float*)d_in[1];
  const float* hebb0 = (const float*)d_in[2];
  const float* Wz    = (const float*)d_in[3];
  const float* Uz    = (const float*)d_in[4];
  const float* bz    = (const float*)d_in[5];
  const float* Wr    = (const float*)d_in[6];
  const float* Ur    = (const float*)d_in[7];
  const float* br    = (const float*)d_in[8];
  const float* Wh    = (const float*)d_in[9];
  const float* Uh    = (const float*)d_in[10];
  const float* bh    = (const float*)d_in[11];
  const float* Al    = (const float*)d_in[12];
  const float* eta   = (const float*)d_in[13];
  const float* Wo    = (const float*)d_in[14];
  const float* bo    = (const float*)d_in[15];
  const int*   seed  = (const int*)d_in[16];

  // Workspace carve (~107 MB total)
  char* ws = (char*)d_ws;
  size_t o = 0;
  _Float16* X16 = (_Float16*)(ws + o); o += (size_t)T_STEPS * IN_DIM * 2;
  _Float16* W16 = (_Float16*)(ws + o); o += 3ull * IN_DIM * H_DIM * 2;
  _Float16* P16 = (_Float16*)(ws + o); o += 3ull * T_STEPS * H_DIM * 2;
  _Float16* U16 = (_Float16*)(ws + o); o += 4ull * H_DIM * H_DIM * 2;
  float* hebb    = (float*)(ws + o);   o += (size_t)H_DIM * H_DIM * 4;
  float* hbuf    = (float*)(ws + o);   o += 2ull * H_DIM * 4;
  float* part    = (float*)(ws + o);   o += 64ull * H_DIM * 4;
  float* outPart = (float*)(ws + o);   o += (size_t)T_STEPS * OUT_DIM * 4 * 4;
  unsigned* cnt  = (unsigned*)(ws + o);

  float* partZ = part;
  float* partR = part + 16 * H_DIM;
  float* partB = part + 32 * H_DIM;

  const _Float16* projZ = P16;
  const _Float16* projR = P16 + (size_t)T_STEPS * H_DIM;
  const _Float16* projH = P16 + 2ull * T_STEPS * H_DIM;
  const _Float16* Uz16  = U16;
  const _Float16* Ur16  = U16 + (size_t)H_DIM * H_DIM;
  const _Float16* Uh16  = U16 + 2ull * H_DIM * H_DIM;
  const _Float16* Al16  = U16 + 3ull * H_DIM * H_DIM;

  // d_out layout: output[T,4] | action[T] (int) | hT[H] | hebbT[H,H]
  float* out_out = (float*)d_out;
  int*   out_act = (int*)((float*)d_out + T_STEPS * OUT_DIM);
  float* out_ht  = (float*)d_out + T_STEPS * OUT_DIM + T_STEPS;
  float* out_hb  = out_ht + H_DIM;

  k_convert<<<4096, 256, 0, stream>>>(x, Wz, Wr, Wh, Uz, Ur, Uh, Al, hebb0, h0,
                                      X16, W16, U16, hebb, hbuf, cnt);
  k_gemm<<<dim3(H_DIM / 16, T_STEPS / 128, 3), 256, 0, stream>>>(X16, W16, bz, br, bh, P16);
  k_scan<<<NWG, BLK, 0, stream>>>(projZ, projR, projH, Uz16, Ur16, Uh16, Al16,
                                  hebb, hbuf, partZ, partR, partB, outPart, Wo, eta, cnt);
  k_fhebb<<<(H_DIM * H_DIM + 255) / 256, 256, 0, stream>>>(hebb, hbuf, eta, out_hb, out_ht);
  k_post<<<(T_STEPS + 255) / 256, 256, 0, stream>>>(outPart, bo, seed, out_out, out_act);
}
